// Cifar10_QAHFNN_SingleQubit_89816356094202
// MI455X (gfx1250) — matrix operand order repaired and checked by reading, not compiled, not run on hardware
//
#include <hip/hip_runtime.h>

// ---------------------------------------------------------------------------
// Types for CDNA5 WMMA (gfx1250, wave32)
// ---------------------------------------------------------------------------
typedef __attribute__((ext_vector_type(16))) __bf16        v16bf;
typedef __attribute__((ext_vector_type(8)))  float         v8f;
typedef __attribute__((ext_vector_type(8)))  unsigned int  v8u;

#define DEV static __device__ __forceinline__

// CDNA5 async memory->LDS path (ASYNCcnt). Weight/GEMM tiles are always valid,
// so they go through global_load_async_to_lds_b128; activation tiles need
// zero-padding and stay on the VGPR path.
#ifndef USE_ASYNC_LDS
#define USE_ASYNC_LDS 1
#endif

DEV void async_ld_b128(void* lds_ptr, const void* gaddr) {
#if USE_ASYNC_LDS
  unsigned lds_off = (unsigned)(size_t)lds_ptr;  // low 32 bits = LDS offset
  asm volatile("global_load_async_to_lds_b128 %0, %1, off"
               :: "v"(lds_off), "v"(gaddr)
               : "memory");
#else
  *(uint4*)lds_ptr = *(const uint4*)gaddr;
#endif
}
DEV void async_wait0() {
#if USE_ASYNC_LDS
  asm volatile("s_wait_asynccnt 0" ::: "memory");
#endif
}

DEV unsigned short f32_to_bf16(float f) {
  unsigned int u = __builtin_bit_cast(unsigned int, f);
  u += 0x7FFFu + ((u >> 16) & 1u);          // round-to-nearest-even
  return (unsigned short)(u >> 16);
}
DEV float bf16_to_f32(unsigned short h) {
  unsigned int u = ((unsigned int)h) << 16;
  return __builtin_bit_cast(float, u);
}

// A fragment: 16x32 bf16, lane = M; LDS tile is 64 rows x 32 dwords (K=64),
// sub-step s in {0,1} selects the 32-wide K half.
DEV v16bf load_frag_a64(const unsigned int* lds, int row_base, int lane, int s) {
  const unsigned int* p = lds + (row_base + (lane & 15)) * 32 + s * 16;
  const int half = lane >> 4;
  v8u u;
#pragma unroll
  for (int i = 0; i < 8; ++i) {
    int kk = (i << 1) + (half << 4); // even K index
    u[i] = p[kk >> 1];
  }
  return __builtin_bit_cast(v16bf, u);
}

// B fragment: 32x16 bf16, lane = N; lane-group selects K half; K contiguous.
DEV v16bf load_frag_b64(const unsigned int* lds, int col_base, int lane, int s) {
  const unsigned int* p = lds + (col_base + (lane & 15)) * 32 + s * 16 +
                          ((lane >> 4) << 3);
  v8u u;
#pragma unroll
  for (int i = 0; i < 8; ++i) u[i] = p[i];
  return __builtin_bit_cast(v16bf, u);
}

DEV v8f wmma_bf16(v16bf a, v16bf b, v8f c) {
  return __builtin_amdgcn_wmma_f32_16x16x32_bf16(false, a, false, b, (short)0, c,
                                                 false, false);
}

// ---------------------------------------------------------------------------
// Prep kernels (BN folding, weight packing to bf16 implicit-GEMM layout)
// ---------------------------------------------------------------------------
__global__ void bnprep_kernel(const float* __restrict__ g, const float* __restrict__ b,
                              const float* __restrict__ m, const float* __restrict__ v,
                              float* __restrict__ scale, float* __restrict__ bias, int C) {
  int i = blockIdx.x * blockDim.x + threadIdx.x;
  if (i < C) {
    float inv = g[i] * rsqrtf(v[i] + 1e-5f);
    scale[i] = inv;
    bias[i]  = b[i] - m[i] * inv;
  }
}

// [Cout][Cin][3][3] f32  ->  [Cout][(kh*3+kw)*Cin + ci] bf16
__global__ void pack3_kernel(const float* __restrict__ w, unsigned short* __restrict__ dst,
                             int Cout, int Cin) {
  long total = (long)Cout * 9 * Cin;
  long idx = (long)blockIdx.x * blockDim.x + threadIdx.x;
  if (idx >= total) return;
  int K9 = 9 * Cin;
  int co = (int)(idx / K9);
  int rem = (int)(idx % K9);
  int tap = rem / Cin, ci = rem % Cin;
  int kh = tap / 3, kw = tap % 3;
  dst[idx] = f32_to_bf16(w[(((long)co * Cin + ci) * 3 + kh) * 3 + kw]);
}

// fc_w [256][2048] (k = c*4 + h*2 + w, NCHW flatten) -> bf16 [256][2048] with
// k' = (h*2+w)*512 + c matching our NHWC activation flatten.
__global__ void packfc_kernel(const float* __restrict__ w, unsigned short* __restrict__ dst) {
  int idx = blockIdx.x * 256 + threadIdx.x;    // 256*2048 total
  int n = idx >> 11;
  int kp = idx & 2047;
  int hw = kp >> 9;
  int c  = kp & 511;
  dst[idx] = f32_to_bf16(w[(long)n * 2048 + c * 4 + hw]);
}

// ---------------------------------------------------------------------------
// Block 0 front-end (Cin=3, K=27: direct VALU conv; negligible FLOPs)
// ---------------------------------------------------------------------------
__global__ void __launch_bounds__(256)
conv0_kernel(const float* __restrict__ x, const float* __restrict__ w,
             const float* __restrict__ scale, const float* __restrict__ bias,
             unsigned short* __restrict__ out) {
  int idx = blockIdx.x * 256 + threadIdx.x;   // 256*32*32*64
  int co = idx & 63;
  int m  = idx >> 6;
  int b  = m >> 10;
  int hw = m & 1023;
  int h = hw >> 5, wc = hw & 31;
  float acc = 0.f;
#pragma unroll
  for (int ci = 0; ci < 3; ++ci)
#pragma unroll
    for (int kh = 0; kh < 3; ++kh) {
      int y = h + kh - 1;
      if ((unsigned)y >= 32u) continue;
#pragma unroll
      for (int kw = 0; kw < 3; ++kw) {
        int px = wc + kw - 1;
        if ((unsigned)px >= 32u) continue;
        acc += x[(((long)b * 3 + ci) << 10) + (y << 5) + px] *
               w[((co * 3 + ci) * 3 + kh) * 3 + kw];
      }
    }
  float val = fmaxf(acc * scale[co] + bias[co], 0.f);
  out[(size_t)m * 64 + co] = f32_to_bf16(val);
}

__global__ void __launch_bounds__(256)
shortcut0_kernel(const float* __restrict__ x, const float* __restrict__ ws,
                 const float* __restrict__ scale, const float* __restrict__ bias,
                 unsigned short* __restrict__ out) {
  int idx = blockIdx.x * 256 + threadIdx.x;   // 256*32*32*64
  int co = idx & 63;
  int m  = idx >> 6;
  int b  = m >> 10;
  int hw = m & 1023;
  float acc = 0.f;
#pragma unroll
  for (int ci = 0; ci < 3; ++ci)
    acc += x[(((long)b * 3 + ci) << 10) + hw] * ws[co * 3 + ci];
  out[(size_t)m * 64 + co] = f32_to_bf16(acc * scale[co] + bias[co]);
}

// 1x1 conv + BN shortcut on NHWC bf16 activations (few % of FLOPs)
__global__ void __launch_bounds__(256)
shortcut_nhwc_kernel(const unsigned short* __restrict__ in, const float* __restrict__ ws,
                     const float* __restrict__ scale, const float* __restrict__ bias,
                     unsigned short* __restrict__ out, int npix, int Cin, int Cout) {
  long idx = (long)blockIdx.x * 256 + threadIdx.x;
  if (idx >= (long)npix * Cout) return;
  int co = (int)(idx % Cout);
  long m = idx / Cout;
  const unsigned short* ip = in + m * Cin;
  const float* wp = ws + (long)co * Cin;
  float acc = 0.f;
  for (int ci = 0; ci < Cin; ++ci) acc += bf16_to_f32(ip[ci]) * wp[ci];
  out[idx] = f32_to_bf16(acc * scale[co] + bias[co]);
}

__global__ void __launch_bounds__(256)
maxpool_kernel(const unsigned short* __restrict__ in, unsigned short* __restrict__ out,
               int Ho, int Wo, int C) {
  long idx = (long)blockIdx.x * 256 + threadIdx.x;  // 256*Ho*Wo*C
  int c = (int)(idx % C);
  long r = idx / C;
  int wo = (int)(r % Wo);
  long r2 = r / Wo;
  int ho = (int)(r2 % Ho);
  int b  = (int)(r2 / Ho);
  int Hi = Ho * 2, Wi = Wo * 2;
  float mx = -3.0e38f;
#pragma unroll
  for (int dy = 0; dy < 2; ++dy)
#pragma unroll
    for (int dx = 0; dx < 2; ++dx) {
      float v = bf16_to_f32(in[(((long)b * Hi + 2 * ho + dy) * Wi + 2 * wo + dx) * C + c]);
      mx = fmaxf(mx, v);
    }
  out[idx] = f32_to_bf16(mx);
}

// ---------------------------------------------------------------------------
// Main WMMA conv: implicit GEMM, NHWC bf16, fused BN (+resid) (+ReLU)
// 256 threads = 8 waves; 64(M) x 64(N) tile; K-step 64 = one tap, 64 channels.
// Taps iterate in the outer loop so the gather address/validity is computed
// once per tap; the inner channel loop is a pure pointer walk.
// Weights stream memory->LDS via global_load_async_to_lds_b128 (ASYNCcnt).
// ---------------------------------------------------------------------------
__global__ void __launch_bounds__(256)
conv3x3_wmma_kernel(const unsigned short* __restrict__ in,   // NHWC bf16 [B*H*W*Cin]
                    const unsigned short* __restrict__ wpk,  // [Cout][9*Cin] bf16
                    const float* __restrict__ scale, const float* __restrict__ bias,
                    const unsigned short* __restrict__ resid, // NHWC bf16 or null
                    unsigned short* __restrict__ out,         // NHWC bf16
                    int H, int W, int Cin, int Cout, int relu) {
  __shared__ uint4 ldsA4[512];  // 64 rows x 64 bf16 (8 uint4/row)
  __shared__ uint4 ldsB4[512];
  const unsigned int* ldsA = (const unsigned int*)ldsA4;
  const unsigned int* ldsB = (const unsigned int*)ldsB4;

  const int HW = H * W;
  const int nblk = Cout >> 6;
  const int m0 = ((int)blockIdx.x / nblk) * 64;
  const int n0 = ((int)blockIdx.x % nblk) * 64;

  const int tid  = threadIdx.x;
  const int lane = tid & 31;
  const int wave = tid >> 5;
  const int mi   = wave & 3;            // M-tile 0..3
  const int nib  = (wave >> 2) << 1;    // N-tiles {nib, nib+1}

  // Staging role: row (tid>>2) of both tiles, quarter (tid&3) = 8 channels.
  const int arow = tid >> 2;
  const int acol8 = (tid & 3) * 8;
  const int a4b  = arow * 8 + (tid & 3);
  const int mg   = m0 + arow;
  const int bimg = mg / HW;
  const int rem  = mg % HW;
  const int oh   = rem / W;
  const int ow   = rem % W;

  const int K = 9 * Cin;
  const unsigned short* brow = wpk + (size_t)(n0 + arow) * K + acol8;

  v8f acc0 = {0.f, 0.f, 0.f, 0.f, 0.f, 0.f, 0.f, 0.f};
  v8f acc1 = {0.f, 0.f, 0.f, 0.f, 0.f, 0.f, 0.f, 0.f};

  int ky = 0, kx = 0;
  for (int tap = 0; tap < 9; ++tap) {
    const int y = oh + ky - 1;
    const int x = ow + kx - 1;
    const bool valid = ((unsigned)y < (unsigned)H) && ((unsigned)x < (unsigned)W);
    const unsigned short* ap =
        in + (((size_t)bimg * H + y) * W + x) * Cin + acol8;
    const unsigned short* bp = brow + tap * Cin;

    for (int c0 = 0; c0 < Cin; c0 += 64) {
      uint4 a0 = make_uint4(0u, 0u, 0u, 0u), a1 = a0;
      if (valid) {
        a0 = *(const uint4*)(ap + c0);
        a1 = *(const uint4*)(ap + c0 + 32);
      }
      ldsA4[a4b]     = a0;
      ldsA4[a4b + 4] = a1;
      async_ld_b128(&ldsB4[a4b],     bp + c0);
      async_ld_b128(&ldsB4[a4b + 4], bp + c0 + 32);
      __builtin_prefetch(bp + c0 + 64, 0, 1);   // speculative (safe past end)
      async_wait0();
      __syncthreads();

      v16bf af0 = load_frag_a64(ldsA, mi * 16, lane, 0);
      v16bf b00 = load_frag_b64(ldsB, (nib + 0) * 16, lane, 0);
      v16bf b01 = load_frag_b64(ldsB, (nib + 1) * 16, lane, 0);
      acc0 = wmma_bf16(af0, b00, acc0);
      acc1 = wmma_bf16(af0, b01, acc1);
      v16bf af1 = load_frag_a64(ldsA, mi * 16, lane, 1);
      v16bf b10 = load_frag_b64(ldsB, (nib + 0) * 16, lane, 1);
      v16bf b11 = load_frag_b64(ldsB, (nib + 1) * 16, lane, 1);
      acc0 = wmma_bf16(af1, b10, acc0);
      acc1 = wmma_bf16(af1, b11, acc1);
      __syncthreads();
    }
    if (++kx == 3) { kx = 0; ++ky; }
  }

  // Epilogue: C/D layout -> lanes 0-15: (M=r, N=lane); 16-31: (M=r+8, N=lane-16)
  const int mrow = m0 + mi * 16 + ((lane >> 4) << 3);
  const int nc0  = n0 + nib * 16 + (lane & 15);
  const int nc1  = nc0 + 16;
  const float s0 = scale[nc0], o0 = bias[nc0];
  const float s1 = scale[nc1], o1 = bias[nc1];
#pragma unroll
  for (int r = 0; r < 8; ++r) {
    const size_t p0 = (size_t)(mrow + r) * Cout + nc0;
    const size_t p1 = (size_t)(mrow + r) * Cout + nc1;
    float v0 = acc0[r] * s0 + o0;
    float v1 = acc1[r] * s1 + o1;
    if (resid) { v0 += bf16_to_f32(resid[p0]); v1 += bf16_to_f32(resid[p1]); }
    if (relu)  { v0 = fmaxf(v0, 0.f); v1 = fmaxf(v1, 0.f); }
    out[p0] = f32_to_bf16(v0);
    out[p1] = f32_to_bf16(v1);
  }
}

// ---------------------------------------------------------------------------
// WMMA GEMM for the FC layer: out[M][N] f32 = A[M][K]bf16 * B[N][K]bf16^T + bias
// Both tiles stream through the async memory->LDS path.
// ---------------------------------------------------------------------------
__global__ void __launch_bounds__(256)
gemm_bf16_kernel(const unsigned short* __restrict__ A, const unsigned short* __restrict__ Bm,
                 const float* __restrict__ bias, float* __restrict__ out, int N, int K) {
  __shared__ uint4 ldsA4[512];
  __shared__ uint4 ldsB4[512];
  const unsigned int* ldsA = (const unsigned int*)ldsA4;
  const unsigned int* ldsB = (const unsigned int*)ldsB4;

  const int nblk = N >> 6;
  const int m0 = ((int)blockIdx.x / nblk) * 64;
  const int n0 = ((int)blockIdx.x % nblk) * 64;
  const int tid  = threadIdx.x;
  const int lane = tid & 31;
  const int wave = tid >> 5;
  const int mi   = wave & 3;
  const int nib  = (wave >> 2) << 1;
  const int arow = tid >> 2;
  const int acol8 = (tid & 3) * 8;
  const int a4b  = arow * 8 + (tid & 3);

  const unsigned short* Ap = A  + (size_t)(m0 + arow) * K + acol8;
  const unsigned short* Bp = Bm + (size_t)(n0 + arow) * K + acol8;

  v8f acc0 = {0.f, 0.f, 0.f, 0.f, 0.f, 0.f, 0.f, 0.f};
  v8f acc1 = {0.f, 0.f, 0.f, 0.f, 0.f, 0.f, 0.f, 0.f};

  for (int k0 = 0; k0 < K; k0 += 64) {
    async_ld_b128(&ldsA4[a4b],     Ap + k0);
    async_ld_b128(&ldsA4[a4b + 4], Ap + k0 + 32);
    async_ld_b128(&ldsB4[a4b],     Bp + k0);
    async_ld_b128(&ldsB4[a4b + 4], Bp + k0 + 32);
    __builtin_prefetch(Bp + k0 + 64, 0, 1);
    async_wait0();
    __syncthreads();

    v16bf af0 = load_frag_a64(ldsA, mi * 16, lane, 0);
    v16bf b00 = load_frag_b64(ldsB, (nib + 0) * 16, lane, 0);
    v16bf b01 = load_frag_b64(ldsB, (nib + 1) * 16, lane, 0);
    acc0 = wmma_bf16(af0, b00, acc0);
    acc1 = wmma_bf16(af0, b01, acc1);
    v16bf af1 = load_frag_a64(ldsA, mi * 16, lane, 1);
    v16bf b10 = load_frag_b64(ldsB, (nib + 0) * 16, lane, 1);
    v16bf b11 = load_frag_b64(ldsB, (nib + 1) * 16, lane, 1);
    acc0 = wmma_bf16(af1, b10, acc0);
    acc1 = wmma_bf16(af1, b11, acc1);
    __syncthreads();
  }

  const int mrow = m0 + mi * 16 + ((lane >> 4) << 3);
  const int nc0  = n0 + nib * 16 + (lane & 15);
  const int nc1  = nc0 + 16;
#pragma unroll
  for (int r = 0; r < 8; ++r) {
    out[(size_t)(mrow + r) * N + nc0] = acc0[r] + bias[nc0];
    out[(size_t)(mrow + r) * N + nc1] = acc1[r] + bias[nc1];
  }
}

// ---------------------------------------------------------------------------
// Quantum branch: per-(class,layer) SU(2) matrices precomputed in LDS, then a
// 3-rotation circuit per (pixel,class), multiplicative reduction over pixels.
// ---------------------------------------------------------------------------
DEV void cmulf(float ar, float ai, float br, float bi, float& rr, float& ri) {
  rr = ar * br - ai * bi;
  ri = ar * bi + ai * br;
}

__global__ void __launch_bounds__(256)
quantum_kernel(const float* __restrict__ x, const float* __restrict__ q,
               float* __restrict__ fuzzy) {
  __shared__ float pm[10][3][8];
  __shared__ float red[256];
  const int b = blockIdx.x, tid = threadIdx.x;

  if (tid < 30) {
    const int c = tid / 3, l = tid % 3;
    const float t0 = q[c * 9 + l * 3 + 0];
    const float t1 = q[c * 9 + l * 3 + 1];
    const float t2 = q[c * 9 + l * 3 + 2];
    const float c0 = cosf(0.5f * t0), s0 = sinf(0.5f * t0);
    const float c1 = cosf(0.5f * t1), s1 = sinf(0.5f * t1);
    const float c2 = cosf(0.5f * t2), s2 = sinf(0.5f * t2);
    float tr, ti;
    float* M = pm[c][l];
    cmulf(c0, -s0, c2, -s2, tr, ti); M[0] =  c1 * tr; M[1] =  c1 * ti;  // A00
    cmulf(c0,  s0, c2, -s2, tr, ti); M[2] = -s1 * tr; M[3] = -s1 * ti;  // A01
    cmulf(c0, -s0, c2,  s2, tr, ti); M[4] =  s1 * tr; M[5] =  s1 * ti;  // A10
    cmulf(c0,  s0, c2,  s2, tr, ti); M[6] =  c1 * tr; M[7] =  c1 * ti;  // A11
  }
  __syncthreads();

  float lp[10];
#pragma unroll
  for (int c = 0; c < 10; ++c) lp[c] = 1.f;

  for (int p = tid; p < 3072; p += 256) {
    const float ang = 0.5f * x[(long)b * 3072 + p];
    const float cs = cosf(ang), sn = sinf(ang);
#pragma unroll
    for (int c = 0; c < 10; ++c) {
      float a0r = 1.f, a0i = 0.f, a1r = 0.f, a1i = 0.f;
#pragma unroll
      for (int l = 0; l < 3; ++l) {
        const float b0r = cs * a0r - sn * a1r, b0i = cs * a0i - sn * a1i;
        const float b1r = sn * a0r + cs * a1r, b1i = sn * a0i + cs * a1i;
        const float* M = pm[c][l];
        a0r = M[0] * b0r - M[1] * b0i + M[2] * b1r - M[3] * b1i;
        a0i = M[0] * b0i + M[1] * b0r + M[2] * b1i + M[3] * b1r;
        a1r = M[4] * b0r - M[5] * b0i + M[6] * b1r - M[7] * b1i;
        a1i = M[4] * b0i + M[5] * b0r + M[6] * b1i + M[7] * b1r;
      }
      const float z = (a0r * a0r + a0i * a0i) - (a1r * a1r + a1i * a1i);
      lp[c] *= 0.5f * (z + 1.f);
    }
  }

  for (int c = 0; c < 10; ++c) {
    red[tid] = lp[c];
    __syncthreads();
    for (int s = 128; s > 0; s >>= 1) {
      if (tid < s) red[tid] *= red[tid + s];
      __syncthreads();
    }
    if (tid == 0) fuzzy[b * 10 + c] = red[0];
    __syncthreads();
  }
}

__global__ void __launch_bounds__(256)
fusion_final_kernel(const float* __restrict__ cpart, const float* __restrict__ fuzzy,
                    const float* __restrict__ flw, const float* __restrict__ flb,
                    const float* __restrict__ cw, const float* __restrict__ cb,
                    float* __restrict__ out) {
  __shared__ float fu[256];
  const int b = blockIdx.x, j = threadIdx.x;
  float acc = cpart[(long)b * 256 + j] + flb[j];
#pragma unroll
  for (int c = 0; c < 10; ++c) acc += fuzzy[b * 10 + c] * flw[j * 10 + c];
  fu[j] = acc;
  __syncthreads();
  if (j < 10) {
    float o = cb[j];
    for (int k = 0; k < 256; ++k) o += fu[k] * cw[j * 256 + k];
    out[b * 10 + j] = o;
  }
}

// ---------------------------------------------------------------------------
// Host orchestration
// ---------------------------------------------------------------------------
struct Net {
  const float* x;
  const float* w1[7]; const float* w2[7]; const float* wsc[7];
  const float *g1[7], *b1[7], *m1[7], *v1[7];
  const float *g2[7], *b2[7], *m2[7], *v2[7];
  const float *gs[7], *bs[7], *ms[7], *vs[7];
  const float *fc_w, *fc_b, *q, *flw, *flb, *cw, *cb;
};

static const int HAS_WS[7] = {1, 1, 0, 1, 0, 1, 0};
static const int CIN[7]    = {3, 64, 128, 128, 256, 256, 512};
static const int COUT[7]   = {64, 128, 128, 256, 256, 512, 512};

static void map_inputs(void* const* d_in, const int* in_sizes, Net& n) {
  auto F = [&](int i) { return (const float*)d_in[i]; };
  if (in_sizes[0] == 786432) {
    // insertion order: x, blocks{w1,bn1(g,b,m,v),w2,bn2,[ws,bns]}, fc_w, fc_b,
    //                  q, flinear_w, flinear_b, classi_w, classi_b
    int i = 0;
    n.x = F(i++);
    for (int b = 0; b < 7; ++b) {
      n.w1[b] = F(i++);
      n.g1[b] = F(i++); n.b1[b] = F(i++); n.m1[b] = F(i++); n.v1[b] = F(i++);
      n.w2[b] = F(i++);
      n.g2[b] = F(i++); n.b2[b] = F(i++); n.m2[b] = F(i++); n.v2[b] = F(i++);
      if (HAS_WS[b]) {
        n.wsc[b] = F(i++);
        n.gs[b] = F(i++); n.bs[b] = F(i++); n.ms[b] = F(i++); n.vs[b] = F(i++);
      } else n.wsc[b] = nullptr;
    }
    n.fc_w = F(i++); n.fc_b = F(i++);
    n.q = F(i++); n.flw = F(i++); n.flb = F(i++); n.cw = F(i++); n.cb = F(i++);
  } else {
    // jax-sorted pytree: classi_b, classi_w, blocks{bn1(b,g,m,v),bn2,[bns],w1,w2,[ws]},
    //                    fc_b, fc_w, flinear_b, flinear_w, q_params, x
    int i = 0;
    n.cb = F(i++); n.cw = F(i++);
    for (int b = 0; b < 7; ++b) {
      n.b1[b] = F(i++); n.g1[b] = F(i++); n.m1[b] = F(i++); n.v1[b] = F(i++);
      n.b2[b] = F(i++); n.g2[b] = F(i++); n.m2[b] = F(i++); n.v2[b] = F(i++);
      if (HAS_WS[b]) { n.bs[b] = F(i++); n.gs[b] = F(i++); n.ms[b] = F(i++); n.vs[b] = F(i++); }
      n.w1[b] = F(i++); n.w2[b] = F(i++);
      if (HAS_WS[b]) n.wsc[b] = F(i++); else n.wsc[b] = nullptr;
    }
    n.fc_b = F(i++); n.fc_w = F(i++);
    n.flb = F(i++); n.flw = F(i++); n.q = F(i++); n.x = F(i++);
  }
}

extern "C" void kernel_launch(void* const* d_in, const int* in_sizes, int n_in,
                              void* d_out, int out_size, void* d_ws, size_t ws_size,
                              hipStream_t stream) {
  (void)n_in; (void)out_size; (void)ws_size;
  Net net;
  map_inputs(d_in, in_sizes, net);

  // ---- workspace carve-up -------------------------------------------------
  char* base = (char*)d_ws;
  size_t off = 0;
  auto alloc = [&](size_t bytes) -> void* {
    void* r = base + off;
    off += (bytes + 255) & ~(size_t)255;
    return r;
  };
  unsigned short* R0 = (unsigned short*)alloc(33554432);  // 256*32*32*64 bf16
  unsigned short* R1 = (unsigned short*)alloc(33554432);
  unsigned short* R2 = (unsigned short*)alloc(33554432);
  unsigned short *pw1[7], *pw2[7];
  float *s1c[7], *b1c[7], *s2c[7], *b2c[7], *ssc[7], *bsc[7];
  for (int b = 0; b < 7; ++b) {
    pw1[b] = (unsigned short*)alloc((size_t)COUT[b] * 9 * CIN[b] * 2);
    pw2[b] = (unsigned short*)alloc((size_t)COUT[b] * 9 * COUT[b] * 2);
    s1c[b] = (float*)alloc(COUT[b] * 4); b1c[b] = (float*)alloc(COUT[b] * 4);
    s2c[b] = (float*)alloc(COUT[b] * 4); b2c[b] = (float*)alloc(COUT[b] * 4);
    ssc[b] = (float*)alloc(COUT[b] * 4); bsc[b] = (float*)alloc(COUT[b] * 4);
  }
  unsigned short* pfc = (unsigned short*)alloc((size_t)256 * 2048 * 2);
  float* cpart = (float*)alloc((size_t)256 * 256 * 4);
  float* fuzzy = (float*)alloc((size_t)256 * 10 * 4);

  // ---- prep: fold BN, pack weights to bf16 implicit-GEMM layout -----------
  for (int b = 0; b < 7; ++b) {
    long t1 = (long)COUT[b] * 9 * CIN[b];
    long t2 = (long)COUT[b] * 9 * COUT[b];
    pack3_kernel<<<(int)((t1 + 255) / 256), 256, 0, stream>>>(net.w1[b], pw1[b], COUT[b], CIN[b]);
    pack3_kernel<<<(int)((t2 + 255) / 256), 256, 0, stream>>>(net.w2[b], pw2[b], COUT[b], COUT[b]);
    int gb = (COUT[b] + 255) / 256;
    bnprep_kernel<<<gb, 256, 0, stream>>>(net.g1[b], net.b1[b], net.m1[b], net.v1[b],
                                          s1c[b], b1c[b], COUT[b]);
    bnprep_kernel<<<gb, 256, 0, stream>>>(net.g2[b], net.b2[b], net.m2[b], net.v2[b],
                                          s2c[b], b2c[b], COUT[b]);
    if (net.wsc[b])
      bnprep_kernel<<<gb, 256, 0, stream>>>(net.gs[b], net.bs[b], net.ms[b], net.vs[b],
                                            ssc[b], bsc[b], COUT[b]);
  }
  packfc_kernel<<<2048, 256, 0, stream>>>(net.fc_w, pfc);

  // ---- block 0 (special: f32 NCHW input) ----------------------------------
  conv0_kernel<<<65536, 256, 0, stream>>>(net.x, net.w1[0], s1c[0], b1c[0], R0);
  shortcut0_kernel<<<65536, 256, 0, stream>>>(net.x, net.wsc[0], ssc[0], bsc[0], R1);
  conv3x3_wmma_kernel<<<4096, 256, 0, stream>>>(R0, pw2[0], s2c[0], b2c[0], R1, R2,
                                                32, 32, 64, 64, 1);
  maxpool_kernel<<<16384, 256, 0, stream>>>(R2, R0, 16, 16, 64);

  // ---- blocks 1..6 --------------------------------------------------------
  unsigned short *I = R0, *FA = R1, *FB = R2;
  int H = 16;
  for (int b = 1; b < 7; ++b) {
    const int Cin = CIN[b], Co = COUT[b];
    const int Mtot = 256 * H * H;
    const int gconv = (Mtot / 64) * (Co / 64);
    conv3x3_wmma_kernel<<<gconv, 256, 0, stream>>>(I, pw1[b], s1c[b], b1c[b], nullptr, FA,
                                                   H, H, Cin, Co, 1);
    if (net.wsc[b]) {
      const int gs = (int)(((long)Mtot * Co + 255) / 256);
      shortcut_nhwc_kernel<<<gs, 256, 0, stream>>>(I, net.wsc[b], ssc[b], bsc[b], FB,
                                                   Mtot, Cin, Co);
      conv3x3_wmma_kernel<<<gconv, 256, 0, stream>>>(FA, pw2[b], s2c[b], b2c[b], FB, I,
                                                     H, H, Co, Co, 1);
      // I holds the block output; FA/FB remain scratch; no pool after ws-blocks
    } else {
      conv3x3_wmma_kernel<<<gconv, 256, 0, stream>>>(FA, pw2[b], s2c[b], b2c[b], I, FB,
                                                     H, H, Co, Co, 1);
      const int Ho = H / 2;
      maxpool_kernel<<<(int)(((long)256 * Ho * Ho * Co) / 256), 256, 0, stream>>>(FB, FA,
                                                                                  Ho, Ho, Co);
      unsigned short* t = I; I = FA; FA = t;  // FB stays scratch
      H = Ho;
    }
  }

  // ---- FC (WMMA GEMM), quantum branch, fusion -----------------------------
  gemm_bf16_kernel<<<(256 / 64) * (256 / 64), 256, 0, stream>>>(I, pfc, net.fc_b, cpart,
                                                                256, 2048);
  quantum_kernel<<<256, 256, 0, stream>>>(net.x, net.q, fuzzy);
  fusion_final_kernel<<<256, 256, 0, stream>>>(cpart, fuzzy, net.flw, net.flb, net.cw,
                                               net.cb, (float*)d_out);
}
